// SelfMultiheadAttention_67250597921315
// MI455X (gfx1250) — compile-verified, matrix operand order checked by reading, not run on hardware
//
#include <hip/hip_runtime.h>
#include <hip/hip_bf16.h>

typedef __attribute__((ext_vector_type(16))) __bf16 v16bf;
typedef __attribute__((ext_vector_type(8)))  __bf16 v8bf;
typedef __attribute__((ext_vector_type(8)))  float  v8f;

#define WMMA_BF16(a, b, c) \
  __builtin_amdgcn_wmma_f32_16x16x32_bf16(false, (a), false, (b), (short)0, (c), false, false)

// Sizes for this problem
#define NN   1024          // n_node
#define GG   8             // n_graph
#define EE   768           // embed_dim
#define HH   12            // heads
#define DD   64            // head_dim
#define BB   (GG * HH)     // 96 attention batches
#define MM   (NN * GG)     // 8192 rows for the projections
#define K3E  (3 * EE)      // 2304

// Load a 16x32 bf16 fragment (A layout; also B layout when `base` is B^T
// row-major).  Lane l<16 supplies row (row0+l), K = k0 + hi*8 + {0..7} and
// k0 + 16 + hi*8 + {0..7}  -> two contiguous 16-byte loads.
__device__ __forceinline__ v16bf load_frag(const __bf16* base, int ld,
                                           int row0, int k0, int lane) {
  int l  = lane & 15;
  int hi = lane >> 4;
  const __bf16* p = base + (size_t)(row0 + l) * ld + k0 + hi * 8;
  union { v16bf v; v8bf h[2]; } u;
  u.h[0] = *(const v8bf*)p;
  u.h[1] = *(const v8bf*)(p + 16);
  return u.v;
}

// ---------------------------------------------------------------- stage 0
__global__ void cvt_bf16_kernel(const float* __restrict__ in,
                                __bf16* __restrict__ out, int n) {
  int i = blockIdx.x * blockDim.x + threadIdx.x;
  if (i < n) out[i] = (__bf16)in[i];
}

// ---------------------------------------------------------------- stage 1
// qkv = query @ w_in^T + b_in, scattered into head-major Q (scaled), K, V^T.
__global__ void qkv_gemm_kernel(const __bf16* __restrict__ A,   // (8192,768)
                                const __bf16* __restrict__ Bw,  // (2304,768)
                                const float*  __restrict__ bias,
                                __bf16* __restrict__ Qb,        // (96,1024,64)
                                __bf16* __restrict__ Kb,        // (96,1024,64)
                                __bf16* __restrict__ Vt) {      // (96,64,1024)
  int lane = threadIdx.x & 31;
  int wave = threadIdx.x >> 5;
  int col0 = blockIdx.x * 16;                      // 144 col tiles
  int row0 = (blockIdx.y * 8 + wave) * 16;         // 64 row groups * 8 waves
  v8f acc = {};
#pragma unroll 4
  for (int k0 = 0; k0 < EE; k0 += 32) {
    v16bf a = load_frag(A,  EE, row0, k0, lane);
    v16bf b = load_frag(Bw, EE, col0, k0, lane);
    acc = WMMA_BF16(a, b, acc);
  }
  int l = lane & 15, hi = lane >> 4;
  int col    = col0 + l;
  int part   = col / EE;           // 0=q, 1=k, 2=v (uniform per wave)
  int within = col % EE;
  int h = within / DD, d = within % DD;
  float bv = bias[col];
  const float scaling = 0.125f;    // 64^-0.5
#pragma unroll
  for (int r = 0; r < 8; ++r) {
    int row = row0 + r + 8 * hi;   // row = n*G + g
    int n = row >> 3, g = row & 7;
    int bh = g * HH + h;
    float v = acc[r] + bv;
    if (part == 0)
      Qb[((size_t)bh * NN + n) * DD + d] = (__bf16)(v * scaling);
    else if (part == 1)
      Kb[((size_t)bh * NN + n) * DD + d] = (__bf16)v;
    else
      Vt[((size_t)bh * DD + d) * NN + n] = (__bf16)v;
  }
}

// ---------------------------------------------------------------- stage 2
// Flash attention: one wave per 16-row block, 32 key columns per step.
__global__ void attn_kernel(const __bf16* __restrict__ Qb,
                            const __bf16* __restrict__ Kb,
                            const __bf16* __restrict__ Vt,
                            const float*  __restrict__ attn_bias,
                            __bf16* __restrict__ attnB) {       // (N,G,E) bf16
  __shared__ __bf16 lds_p[8][16 * 32];          // per-wave P staging (8 KB)
  int lane = threadIdx.x & 31;
  int wave = threadIdx.x >> 5;
  int l = lane & 15, hi = lane >> 4;
  int b    = blockIdx.x;                        // 96 batches
  int row0 = blockIdx.y * 128 + wave * 16;      // 8 row groups of 128

  const __bf16* Q  = Qb + (size_t)b * NN * DD;
  const __bf16* Kk = Kb + (size_t)b * NN * DD;
  const __bf16* V  = Vt + (size_t)b * DD * NN;
  const float*  Bb = attn_bias + (size_t)b * NN * NN;

  v16bf qf0 = load_frag(Q, DD, row0, 0,  lane);   // Q rows, K=0..31
  v16bf qf1 = load_frag(Q, DD, row0, 32, lane);   // Q rows, K=32..63

  v8f o[4] = {v8f{}, v8f{}, v8f{}, v8f{}};
  float rmax[8], rsum[8];
#pragma unroll
  for (int r = 0; r < 8; ++r) { rmax[r] = -3.0e38f; rsum[r] = 0.0f; }

  for (int j = 0; j < NN; j += 32) {
    // ---- scores: S(16x32) = Q(16x64) @ K^T, two 16x16 tiles
    v8f s0 = {}, s1 = {};
    s0 = WMMA_BF16(qf0, load_frag(Kk, DD, j,      0,  lane), s0);
    s0 = WMMA_BF16(qf1, load_frag(Kk, DD, j,      32, lane), s0);
    s1 = WMMA_BF16(qf0, load_frag(Kk, DD, j + 16, 0,  lane), s1);
    s1 = WMMA_BF16(qf1, load_frag(Kk, DD, j + 16, 32, lane), s1);

    // ---- bias add directly in C-fragment layout: elem(r) -> (row0+r+8*hi, j+l[+16])
    const float* bp = Bb + (size_t)(row0 + hi * 8) * NN + j + l;
#pragma unroll
    for (int r = 0; r < 8; ++r) {
      s0[r] += bp[(size_t)r * NN];
      s1[r] += bp[(size_t)r * NN + 16];
    }

    // ---- online softmax (row stats live across the 16-lane halves)
    float alpha[8];
#pragma unroll
    for (int r = 0; r < 8; ++r) {
      float m = fmaxf(s0[r], s1[r]);
      for (int msk = 1; msk <= 8; msk <<= 1) m = fmaxf(m, __shfl_xor(m, msk));
      float nm = fmaxf(rmax[r], m);
      alpha[r] = __expf(rmax[r] - nm);
      rmax[r]  = nm;
      s0[r] = __expf(s0[r] - nm);
      s1[r] = __expf(s1[r] - nm);
      float ps = s0[r] + s1[r];
      for (int msk = 1; msk <= 8; msk <<= 1) ps += __shfl_xor(ps, msk);
      rsum[r] = rsum[r] * alpha[r] + ps;
    }
#pragma unroll
    for (int t = 0; t < 4; ++t)
#pragma unroll
      for (int r = 0; r < 8; ++r) o[t][r] *= alpha[r];

    // ---- stage P to LDS (C layout -> row major), reload as A fragment
    __bf16* pb = lds_p[wave];
#pragma unroll
    for (int r = 0; r < 8; ++r) {
      pb[(r + 8 * hi) * 32 + l]      = (__bf16)s0[r];
      pb[(r + 8 * hi) * 32 + 16 + l] = (__bf16)s1[r];
    }
    asm volatile("s_wait_dscnt 0" ::: "memory");   // wave-internal cross-lane LDS dep
    union { v16bf v; v8bf h[2]; } up;
    const __bf16* pl = pb + l * 32 + hi * 8;
    up.h[0] = *(const v8bf*)pl;
    up.h[1] = *(const v8bf*)(pl + 16);
    v16bf pf = up.v;

    // ---- O(16x64) += P(16x32) @ V(32x64); V^T rows give contiguous B frags
#pragma unroll
    for (int t = 0; t < 4; ++t)
      o[t] = WMMA_BF16(pf, load_frag(V, NN, t * 16, j, lane), o[t]);
  }

  // ---- normalize + scatter back to (n, g, h*64+d) as bf16
  int g = b / HH, h = b % HH;
#pragma unroll
  for (int r = 0; r < 8; ++r) rsum[r] = 1.0f / rsum[r];
#pragma unroll
  for (int t = 0; t < 4; ++t) {
    int d = t * 16 + l;
#pragma unroll
    for (int r = 0; r < 8; ++r) {
      int n = row0 + r + 8 * hi;
      attnB[((size_t)n * GG + g) * EE + h * DD + d] = (__bf16)(o[t][r] * rsum[r]);
    }
  }
}

// ---------------------------------------------------------------- stage 3
// out = attn @ w_out^T + b_out  (f32 output)
__global__ void out_gemm_kernel(const __bf16* __restrict__ A,   // (8192,768)
                                const __bf16* __restrict__ Bw,  // (768,768)
                                const float*  __restrict__ bias,
                                float* __restrict__ out) {
  int lane = threadIdx.x & 31;
  int wave = threadIdx.x >> 5;
  int col0 = blockIdx.x * 16;                  // 48 col tiles
  int row0 = (blockIdx.y * 8 + wave) * 16;     // 64 row groups
  v8f acc = {};
#pragma unroll 4
  for (int k0 = 0; k0 < EE; k0 += 32) {
    v16bf a = load_frag(A,  EE, row0, k0, lane);
    v16bf b = load_frag(Bw, EE, col0, k0, lane);
    acc = WMMA_BF16(a, b, acc);
  }
  int l = lane & 15, hi = lane >> 4;
  float bv = bias[col0 + l];
#pragma unroll
  for (int r = 0; r < 8; ++r)
    out[(size_t)(row0 + r + 8 * hi) * EE + col0 + l] = acc[r] + bv;
}

// ----------------------------------------------------------------
extern "C" void kernel_launch(void* const* d_in, const int* in_sizes, int n_in,
                              void* d_out, int out_size, void* d_ws, size_t ws_size,
                              hipStream_t stream) {
  const float* query     = (const float*)d_in[0];   // (1024, 8, 768)
  const float* attn_bias = (const float*)d_in[1];   // (96, 1024, 1024)
  const float* w_in      = (const float*)d_in[2];   // (2304, 768)
  const float* b_in      = (const float*)d_in[3];   // (2304,)
  const float* w_out     = (const float*)d_in[4];   // (768, 768)
  const float* b_out     = (const float*)d_in[5];   // (768,)
  float* out = (float*)d_out;

  size_t off = 0;
  auto wsalloc = [&](size_t bytes) -> void* {
    void* p = (char*)d_ws + off;
    off += (bytes + 255) & ~(size_t)255;
    return p;
  };
  __bf16* qryB  = (__bf16*)wsalloc((size_t)MM * EE * 2);
  __bf16* winB  = (__bf16*)wsalloc((size_t)K3E * EE * 2);
  __bf16* woutB = (__bf16*)wsalloc((size_t)EE * EE * 2);
  __bf16* Qb    = (__bf16*)wsalloc((size_t)BB * NN * DD * 2);
  __bf16* Kb    = (__bf16*)wsalloc((size_t)BB * NN * DD * 2);
  __bf16* Vt    = (__bf16*)wsalloc((size_t)BB * DD * NN * 2);
  __bf16* attnB = (__bf16*)wsalloc((size_t)MM * EE * 2);

  // stage 0: f32 -> bf16 conversions
  {
    int n0 = MM * EE, n1 = K3E * EE, n2 = EE * EE;
    cvt_bf16_kernel<<<(n0 + 255) / 256, 256, 0, stream>>>(query, qryB, n0);
    cvt_bf16_kernel<<<(n1 + 255) / 256, 256, 0, stream>>>(w_in,  winB, n1);
    cvt_bf16_kernel<<<(n2 + 255) / 256, 256, 0, stream>>>(w_out, woutB, n2);
  }
  // stage 1: QKV projection + head scatter
  qkv_gemm_kernel<<<dim3(K3E / 16, MM / 128), 256, 0, stream>>>(
      qryB, winB, b_in, Qb, Kb, Vt);
  // stage 2: flash attention over 96 batches
  attn_kernel<<<dim3(BB, NN / 128), 256, 0, stream>>>(
      Qb, Kb, Vt, attn_bias, attnB);
  // stage 3: output projection
  out_gemm_kernel<<<dim3(EE / 16, MM / 128), 256, 0, stream>>>(
      attnB, woutB, b_out, out);
}
